// AttentionBlock_33122787787552
// MI455X (gfx1250) — matrix operand order repaired and checked by reading, not compiled, not run on hardware
//
#include <hip/hip_runtime.h>
#include <stdint.h>

// ---------------------------------------------------------------------------
// AttentionBlock for MI455X (gfx1250, wave32, WMMA 16x16x32 bf16 / f32 accum)
//   x:[32,256,32,32] -> qkv(1x1 conv) -> full spatial attention -> 1x1 conv
// Compute-bound (~52 GFLOP vs ~250MB HBM traffic): bf16 WMMA everywhere,
// flash-attention so the 32x[1024x1024] score tensor never hits HBM.
// Data movement: TDM tensor_load_to_lds for K/V chunks (TENSORcnt), async
// global->LDS b128 for GEMM tile staging (ASYNCcnt), with guarded fallbacks.
// ---------------------------------------------------------------------------

typedef __attribute__((ext_vector_type(16))) __bf16 bf16x16;
typedef __attribute__((ext_vector_type(8)))  float  f32x8;
typedef __attribute__((ext_vector_type(4)))  unsigned v4u;
typedef __attribute__((ext_vector_type(8)))  int      v8i;
typedef __attribute__((ext_vector_type(4)))  int      v4i;

union Frag16 {
  bf16x16 v;
  unsigned u[8];
};

#define DEV_AS1 __attribute__((address_space(1)))
#define DEV_AS3 __attribute__((address_space(3)))

// generic->AS1 is value preserving; generic->AS3 truncates to the 32-bit LDS
// offset (AMDGPU flat LDS aperture keeps the offset in the low bits).
__device__ __forceinline__ DEV_AS1 v4i* to_glb16(const void* p) {
  return (DEV_AS1 v4i*)(uintptr_t)p;
}
__device__ __forceinline__ DEV_AS3 v4i* to_lds16(const void* p) {
  return (DEV_AS3 v4i*)(unsigned)(uintptr_t)p;
}
__device__ __forceinline__ unsigned lds_off(const void* p) {
  return (unsigned)(uintptr_t)p;
}

#if defined(__has_builtin)
#if __has_builtin(__builtin_amdgcn_global_load_async_to_lds_b128)
#define HAVE_ASYNC_LDS 1
#endif
#if __has_builtin(__builtin_amdgcn_tensor_load_to_lds)
#define HAVE_TDM 1
#endif
#if __has_builtin(__builtin_amdgcn_s_wait_asynccnt)
#define WAIT_ASYNC() __builtin_amdgcn_s_wait_asynccnt(0)
#endif
#if __has_builtin(__builtin_amdgcn_s_wait_tensorcnt)
#define WAIT_TENSOR() __builtin_amdgcn_s_wait_tensorcnt(0)
#endif
#endif
#ifndef WAIT_ASYNC
#define WAIT_ASYNC() asm volatile("s_wait_asynccnt 0" ::: "memory")
#endif
#ifndef WAIT_TENSOR
#define WAIT_TENSOR() asm volatile("s_wait_tensorcnt 0" ::: "memory")
#endif

// 16B global -> LDS copy: async (no VGPR round trip) when available.
// Builtin signature (from compiler diagnostic): (v4i AS1*, v4i AS3*, imm, imm).
__device__ __forceinline__ void cp16(void* lds, const void* glb) {
#ifdef HAVE_ASYNC_LDS
  __builtin_amdgcn_global_load_async_to_lds_b128(to_glb16(glb), to_lds16(lds),
                                                 0, 0);
#else
  *(uint4*)lds = *(const uint4*)glb;
#endif
}
__device__ __forceinline__ void cp_fence() {
#ifdef HAVE_ASYNC_LDS
  WAIT_ASYNC();
#endif
}

#ifdef HAVE_TDM
// TDM 2D tile load: nrows rows of row_len bf16 elements (row pitch row_stride
// elements in global), packed contiguously into LDS at lds_addr.
// D# layout per CDNA5 ISA.
__device__ __forceinline__ void tdm_load_2d(unsigned lds_addr, const void* gp,
                                            unsigned row_len, unsigned nrows,
                                            unsigned row_stride) {
  unsigned long long ga = (unsigned long long)(uintptr_t)gp;
  v4u g0;
  g0[0] = 1u;                                       // count=1 user descriptor
  g0[1] = lds_addr;                                 // lds byte address
  g0[2] = (unsigned)(ga & 0xffffffffu);             // global addr [31:0]
  g0[3] = (unsigned)((ga >> 32) & 0x01ffffffu) | (2u << 30);  // [56:32] + type=2
  v8i g1;
  g1[0] = (int)(1u << 16);                          // data_size=1 (2B), mask=0
  g1[1] = (int)((row_len & 0xffffu) << 16);         // tensor_dim0[15:0]
  g1[2] = (int)(((row_len >> 16) & 0xffffu) |       // tensor_dim0[31:16]
                ((nrows & 0xffffu) << 16));         // tensor_dim1[15:0]
  g1[3] = (int)(((nrows >> 16) & 0xffffu) |         // tensor_dim1[31:16]
                ((row_len & 0xffffu) << 16));       // tile_dim0
  g1[4] = (int)(nrows & 0xffffu);                   // tile_dim1 (tile_dim2=0)
  g1[5] = (int)row_stride;                          // tensor_dim0_stride[31:0]
  g1[6] = 0;                                        // stride hi, dim1_stride lo
  g1[7] = 0;
  v4i z4 = {0, 0, 0, 0};
#if __clang_major__ >= 23
  v8i z8 = {0, 0, 0, 0, 0, 0, 0, 0};
  __builtin_amdgcn_tensor_load_to_lds(g0, g1, z4, z4, z8, 0);
#else
  __builtin_amdgcn_tensor_load_to_lds(g0, g1, z4, z4, 0);
#endif
}
#endif

__device__ __forceinline__ unsigned short f2bf(float f) {
  unsigned u = __float_as_uint(f);
  return (unsigned short)((u + 0x7FFFu + ((u >> 16) & 1u)) >> 16);  // RNE
}

// A-fragment (16x32 bf16) k index for VGPR v, lane-half h (ISA 7.12.2 table)
__device__ __forceinline__ int ka(int v, int h) {
  return (v < 4 ? 2 * v : 16 + 2 * (v - 4)) + 8 * h;
}
// B-fragment (32x16 bf16) k index for VGPR v, lane-half h
__device__ __forceinline__ int kb(int v, int h) { return (v < 4 ? 2 * v : 16 + 2 * (v - 4)) + 8 * h; }

__device__ __forceinline__ f32x8 wmma_bf(bf16x16 a, bf16x16 b, f32x8 c) {
  return __builtin_amdgcn_wmma_f32_16x16x32_bf16(
      /*neg_a=*/false, a, /*neg_b=*/false, b,
      /*c_mod=*/(short)0, c, /*reuse_a=*/false, /*reuse_b=*/false);
}

#define NB 32
#define NC 256
#define NN 1024  // H*W

// ---------------------------------------------------------------------------
// prep: fp32 -> bf16 conversions.  xt is [b][n][c] (c contiguous) so GEMM B
// fragments can be loaded as aligned dword pairs along the K dimension.
// ---------------------------------------------------------------------------
__global__ void __launch_bounds__(256) prep_kernel(
    const float* __restrict__ X, const float* __restrict__ Win,
    const float* __restrict__ Wout, unsigned short* __restrict__ Xt,
    unsigned short* __restrict__ WinB, unsigned short* __restrict__ WoutB) {
  const int stride = gridDim.x * blockDim.x;
  const int tid = blockIdx.x * blockDim.x + threadIdx.x;
  for (int i = tid; i < 3 * NC * NC; i += stride) WinB[i] = f2bf(Win[i]);
  for (int i = tid; i < NC * NC; i += stride) WoutB[i] = f2bf(Wout[i]);
  for (int i = tid; i < NB * NN * NC; i += stride) {
    int c = i & (NC - 1);
    int n = (i >> 8) & (NN - 1);
    int b = i >> 18;
    Xt[i] = f2bf(X[((size_t)b * NC + c) * NN + n]);
  }
}

// ---------------------------------------------------------------------------
// qkv_gemm: QKV[b,o,n] = sum_c Win[o,c]*x[b,c,n] + b_in[o]
// 128x128 block tile, 8 waves (4 o-groups x 2 n-groups), K loop of 32.
// Writes Qt[b][n][c] (scaled by 1/16), Kt[b][n][c], V[b][c][n].
// ---------------------------------------------------------------------------
__global__ void __launch_bounds__(256) qkv_gemm_kernel(
    const unsigned short* __restrict__ Wb,  // [768][256] bf16
    const unsigned short* __restrict__ Xt,  // [B][1024][256] bf16
    const float* __restrict__ bias,         // [768]
    unsigned short* __restrict__ Qt, unsigned short* __restrict__ Kt,
    unsigned short* __restrict__ Vm) {
  __shared__ unsigned short lA[128 * 32];  // [o][c]
  __shared__ unsigned short lB[128 * 32];  // [n][c]
  const int b = blockIdx.z;
  const int oo = blockIdx.y * 128;
  const int nn = blockIdx.x * 128;
  const int t = threadIdx.x;
  const int w = t >> 5, lane = t & 31, hh = lane >> 4, lc = lane & 15;
  const int wo = (w & 3) * 32;   // wave o offset in block tile
  const int wn = (w >> 2) * 64;  // wave n offset in block tile

  f32x8 acc[2][4];
  for (int i = 0; i < 2; ++i)
    for (int j = 0; j < 4; ++j)
      for (int e = 0; e < 8; ++e) acc[i][j][e] = 0.f;

  for (int kc = 0; kc < NC; kc += 32) {
    __syncthreads();
    for (int i = 0; i < 2; ++i) {  // 512 x 16B segments, 256 threads
      int seg = t + i * 256;
      int row = seg >> 2, sc = (seg & 3) * 8;
      cp16(&lA[row * 32 + sc], &Wb[(size_t)(oo + row) * NC + kc + sc]);
      cp16(&lB[row * 32 + sc], &Xt[((size_t)b * NN + nn + row) * NC + kc + sc]);
    }
    cp_fence();
    __syncthreads();

    Frag16 af[2], bfr[4];
#pragma unroll
    for (int ot = 0; ot < 2; ++ot)
#pragma unroll
      for (int v = 0; v < 8; ++v)
        af[ot].u[v] = *(const unsigned*)&lA[(wo + ot * 16 + lc) * 32 + ka(v, hh)];
#pragma unroll
    for (int nt = 0; nt < 4; ++nt)
#pragma unroll
      for (int v = 0; v < 8; ++v)
        bfr[nt].u[v] = *(const unsigned*)&lB[(wn + nt * 16 + lc) * 32 + kb(v, hh)];
#pragma unroll
    for (int ot = 0; ot < 2; ++ot)
#pragma unroll
      for (int nt = 0; nt < 4; ++nt)
        acc[ot][nt] = wmma_bf(af[ot].v, bfr[nt].v, acc[ot][nt]);
  }

  // Epilogue. D layout: row = j + 8*hh, col = lc (ISA 7.12.2).
#pragma unroll
  for (int ot = 0; ot < 2; ++ot) {
    const int ob = oo + wo + ot * 16 + 8 * hh;  // +j gives output channel
#pragma unroll
    for (int nt = 0; nt < 4; ++nt) {
      const int n = nn + wn + nt * 16 + lc;
      float vals[8];
#pragma unroll
      for (int j = 0; j < 8; ++j) vals[j] = acc[ot][nt][j] + bias[ob + j];
      if (ob < NC) {  // Q, pre-scaled by 1/sqrt(C)=1/16
        unsigned short p[8];
#pragma unroll
        for (int j = 0; j < 8; ++j) p[j] = f2bf(vals[j] * 0.0625f);
        *(uint4*)&Qt[((size_t)b * NN + n) * NC + ob] = *(const uint4*)p;
      } else if (ob < 2 * NC) {  // K
        unsigned short p[8];
#pragma unroll
        for (int j = 0; j < 8; ++j) p[j] = f2bf(vals[j]);
        *(uint4*)&Kt[((size_t)b * NN + n) * NC + (ob - NC)] = *(const uint4*)p;
      } else {  // V, stored [c][n] so O-update B-fragments are m-contiguous
#pragma unroll
        for (int j = 0; j < 8; ++j)
          Vm[((size_t)b * NC + (ob - 2 * NC) + j) * NN + n] = f2bf(vals[j]);
      }
    }
  }
}

// ---------------------------------------------------------------------------
// attn: flash attention.  Block = (batch, 128 query rows); each of 8 waves
// privately owns 16 rows: S = Qt*K (4x8 WMMA per 64-key chunk), in-wave
// online softmax (row stats replicated across each 16-lane half), P staged
// through per-wave LDS into A-fragment layout, O += P*V (16x2 WMMA).
// K and V chunks time-share one 32KB LDS buffer, each filled by a single
// TDM 2D-tile descriptor (wave 0) when available.
// ---------------------------------------------------------------------------
__global__ void __launch_bounds__(256, 1) attn_kernel(
    const unsigned short* __restrict__ Qt, const unsigned short* __restrict__ Kt,
    const unsigned short* __restrict__ Vm, unsigned short* __restrict__ Ht) {
  __shared__ unsigned short lKV[64 * 256];  // K phase: [m][c]; V phase: [c][m]
  __shared__ unsigned short lP[8][16 * 64]; // per-wave P tile [row][m]
  const int b = blockIdx.y;
  const int nb = blockIdx.x;
  const int t = threadIdx.x;
  const int w = t >> 5, lane = t & 31, hh = lane >> 4, lc = lane & 15;
  const int n0 = nb * 128 + w * 16;  // wave's 16 query rows

  // Resident Q fragments: 16 rows x 256 c = 8 A-fragments (64 VGPRs bf16).
  Frag16 qF[8];
#pragma unroll
  for (int f = 0; f < 8; ++f)
#pragma unroll
    for (int v = 0; v < 8; ++v)
      qF[f].u[v] =
          *(const unsigned*)&Qt[((size_t)b * NN + n0 + lc) * NC + f * 32 + ka(v, hh)];

  f32x8 oacc[16];  // 16 rows x 256 c accumulator
  for (int ct = 0; ct < 16; ++ct)
    for (int e = 0; e < 8; ++e) oacc[ct][e] = 0.f;
  float mrow[8], srow[8];
  for (int j = 0; j < 8; ++j) { mrow[j] = -1e30f; srow[j] = 0.f; }

  for (int mc = 0; mc < NN / 64; ++mc) {
    const int m0 = mc * 64;
    __syncthreads();  // all waves done reading previous V chunk
#ifdef HAVE_TDM
    if (w == 0) {  // one 2D descriptor moves the whole 32KB K chunk [m][c]
      tdm_load_2d(lds_off(&lKV[0]), &Kt[((size_t)b * NN + m0) * NC],
                  /*row_len=*/NC, /*nrows=*/64, /*row_stride=*/NC);
      WAIT_TENSOR();
    }
#else
    for (int i = 0; i < 8; ++i) {  // stage K chunk [m][c], 2048 x 16B
      int seg = t + i * 256;
      int row = seg >> 5, sc = (seg & 31) * 8;
      cp16(&lKV[row * 256 + sc], &Kt[((size_t)b * NN + m0 + row) * NC + sc]);
    }
    cp_fence();
#endif
    if (mc + 1 < NN / 64)  // warm L2 for the next K chunk (global_prefetch_b8)
      __builtin_prefetch(&Kt[((size_t)b * NN + m0 + 64 + (t >> 2)) * NC], 0, 3);
    __syncthreads();

    // S = Q^T K for this wave's 16 rows x 64 keys
    f32x8 s[4];
    for (int mt = 0; mt < 4; ++mt)
      for (int e = 0; e < 8; ++e) s[mt][e] = 0.f;
#pragma unroll
    for (int mt = 0; mt < 4; ++mt)
#pragma unroll
      for (int f = 0; f < 8; ++f) {
        Frag16 kf;
#pragma unroll
        for (int v = 0; v < 8; ++v)
          kf.u[v] =
              *(const unsigned*)&lKV[(mt * 16 + lc) * 256 + f * 32 + kb(v, hh)];
        s[mt] = wmma_bf(qF[f].v, kf.v, s[mt]);
      }

    // Online softmax; row j+8*hh lives in the 16 lanes of this half.
    float alpha[8];
#pragma unroll
    for (int j = 0; j < 8; ++j) {
      float cm = s[0][j];
      for (int mt = 1; mt < 4; ++mt) cm = fmaxf(cm, s[mt][j]);
      for (int d = 1; d < 16; d <<= 1) cm = fmaxf(cm, __shfl_xor(cm, d, 32));
      float mn = fmaxf(mrow[j], cm);
      alpha[j] = __expf(mrow[j] - mn);
      float cs = 0.f;
      for (int mt = 0; mt < 4; ++mt) {
        float p = __expf(s[mt][j] - mn);
        s[mt][j] = p;
        cs += p;
      }
      for (int d = 1; d < 16; d <<= 1) cs += __shfl_xor(cs, d, 32);
      srow[j] = srow[j] * alpha[j] + cs;
      mrow[j] = mn;
    }
#pragma unroll
    for (int ct = 0; ct < 16; ++ct)
#pragma unroll
      for (int j = 0; j < 8; ++j) oacc[ct][j] *= alpha[j];

    // P (D layout) -> per-wave LDS in row-major [16][64] bf16
#pragma unroll
    for (int mt = 0; mt < 4; ++mt)
#pragma unroll
      for (int j = 0; j < 8; ++j)
        lP[w][(j + 8 * hh) * 64 + mt * 16 + lc] = f2bf(s[mt][j]);

    __syncthreads();  // all waves done reading K; reuse lKV for V
#ifdef HAVE_TDM
    if (w == 0) {  // V chunk [c][m]: 256 rows x 128B, pitch 2KB -> one D#
      tdm_load_2d(lds_off(&lKV[0]), &Vm[(size_t)b * NC * NN + m0],
                  /*row_len=*/64, /*nrows=*/NC, /*row_stride=*/NN);
      WAIT_TENSOR();
    }
#else
    for (int i = 0; i < 8; ++i) {  // stage V chunk [c][m], 2048 x 16B
      int seg = t + i * 256;
      int row = seg >> 3, sc = (seg & 7) * 8;
      cp16(&lKV[row * 64 + sc], &Vm[((size_t)b * NC + row) * NN + m0 + sc]);
    }
    cp_fence();
#endif
    __syncthreads();

    // O += P * V^T : A = P (16x32 bf16 from lP), B = V[c][m] from lKV
    Frag16 pF[2];
#pragma unroll
    for (int kk = 0; kk < 2; ++kk)
#pragma unroll
      for (int v = 0; v < 8; ++v)
        pF[kk].u[v] = *(const unsigned*)&lP[w][lc * 64 + kk * 32 + ka(v, hh)];
#pragma unroll
    for (int ct = 0; ct < 16; ++ct)
#pragma unroll
      for (int kk = 0; kk < 2; ++kk) {
        Frag16 vf;
#pragma unroll
        for (int v = 0; v < 8; ++v)
          vf.u[v] =
              *(const unsigned*)&lKV[(ct * 16 + lc) * 64 + kk * 32 + kb(v, hh)];
        oacc[ct] = wmma_bf(pF[kk].v, vf.v, oacc[ct]);
      }
  }

  // Finalize: H[n][c] = O / rowsum, bf16 for the output GEMM's B operand.
  float rs[8];
  for (int j = 0; j < 8; ++j) rs[j] = 1.f / srow[j];
#pragma unroll
  for (int ct = 0; ct < 16; ++ct)
#pragma unroll
    for (int j = 0; j < 8; ++j) {
      int n = n0 + j + 8 * hh;
      int c = ct * 16 + lc;
      Ht[((size_t)b * NN + n) * NC + c] = f2bf(oacc[ct][j] * rs[j]);
    }
}

// ---------------------------------------------------------------------------
// out_gemm: out[b,o,n] = sum_c Wout[o,c]*H[c,n] + b_out[o] + x[b,o,n]
// ---------------------------------------------------------------------------
__global__ void __launch_bounds__(256) out_gemm_kernel(
    const unsigned short* __restrict__ Wb,  // [256][256] bf16
    const unsigned short* __restrict__ Ht,  // [B][1024][256] bf16
    const float* __restrict__ bias,         // [256]
    const float* __restrict__ X,            // [B][256][1024]
    float* __restrict__ Out) {
  __shared__ unsigned short lA[128 * 32];
  __shared__ unsigned short lB[128 * 32];
  const int b = blockIdx.z;
  const int oo = blockIdx.y * 128;
  const int nn = blockIdx.x * 128;
  const int t = threadIdx.x;
  const int w = t >> 5, lane = t & 31, hh = lane >> 4, lc = lane & 15;
  const int wo = (w & 3) * 32;
  const int wn = (w >> 2) * 64;

  f32x8 acc[2][4];
  for (int i = 0; i < 2; ++i)
    for (int j = 0; j < 4; ++j)
      for (int e = 0; e < 8; ++e) acc[i][j][e] = 0.f;

  for (int kc = 0; kc < NC; kc += 32) {
    __syncthreads();
    for (int i = 0; i < 2; ++i) {
      int seg = t + i * 256;
      int row = seg >> 2, sc = (seg & 3) * 8;
      cp16(&lA[row * 32 + sc], &Wb[(size_t)(oo + row) * NC + kc + sc]);
      cp16(&lB[row * 32 + sc], &Ht[((size_t)b * NN + nn + row) * NC + kc + sc]);
    }
    cp_fence();
    __syncthreads();

    Frag16 af[2], bfr[4];
#pragma unroll
    for (int ot = 0; ot < 2; ++ot)
#pragma unroll
      for (int v = 0; v < 8; ++v)
        af[ot].u[v] = *(const unsigned*)&lA[(wo + ot * 16 + lc) * 32 + ka(v, hh)];
#pragma unroll
    for (int nt = 0; nt < 4; ++nt)
#pragma unroll
      for (int v = 0; v < 8; ++v)
        bfr[nt].u[v] = *(const unsigned*)&lB[(wn + nt * 16 + lc) * 32 + kb(v, hh)];
#pragma unroll
    for (int ot = 0; ot < 2; ++ot)
#pragma unroll
      for (int nt = 0; nt < 4; ++nt)
        acc[ot][nt] = wmma_bf(af[ot].v, bfr[nt].v, acc[ot][nt]);
  }

#pragma unroll
  for (int ot = 0; ot < 2; ++ot) {
    const int ob = oo + wo + ot * 16 + 8 * hh;
#pragma unroll
    for (int nt = 0; nt < 4; ++nt) {
      const int n = nn + wn + nt * 16 + lc;
#pragma unroll
      for (int j = 0; j < 8; ++j) {
        size_t idx = ((size_t)b * NC + ob + j) * NN + n;
        Out[idx] = acc[ot][nt][j] + bias[ob + j] + X[idx];  // bias + residual
      }
    }
  }
}

// ---------------------------------------------------------------------------
extern "C" void kernel_launch(void* const* d_in, const int* in_sizes, int n_in,
                              void* d_out, int out_size, void* d_ws,
                              size_t ws_size, hipStream_t stream) {
  (void)in_sizes; (void)n_in; (void)out_size; (void)ws_size;
  const float* x = (const float*)d_in[0];
  const float* w_in = (const float*)d_in[1];
  const float* b_in = (const float*)d_in[2];
  const float* w_out = (const float*)d_in[3];
  const float* b_out = (const float*)d_in[4];
  float* out = (float*)d_out;

  char* ws = (char*)d_ws;
  size_t off = 0;
  auto take = [&](size_t bytes) -> void* {
    void* p = ws + off;
    off += (bytes + 255) & ~(size_t)255;
    return p;
  };
  unsigned short* WinB = (unsigned short*)take((size_t)3 * NC * NC * 2);
  unsigned short* WoutB = (unsigned short*)take((size_t)NC * NC * 2);
  unsigned short* Xt = (unsigned short*)take((size_t)NB * NN * NC * 2);
  unsigned short* Qt = (unsigned short*)take((size_t)NB * NN * NC * 2);
  unsigned short* Kt = (unsigned short*)take((size_t)NB * NN * NC * 2);
  unsigned short* Vm = (unsigned short*)take((size_t)NB * NC * NN * 2);
  unsigned short* Ht = (unsigned short*)take((size_t)NB * NN * NC * 2);

  prep_kernel<<<2048, 256, 0, stream>>>(x, w_in, w_out, Xt, WinB, WoutB);
  qkv_gemm_kernel<<<dim3(8, 6, NB), 256, 0, stream>>>(WinB, Xt, b_in, Qt, Kt, Vm);
  attn_kernel<<<dim3(8, NB), 256, 0, stream>>>(Qt, Kt, Vm, Ht);
  out_gemm_kernel<<<dim3(8, 2, NB), 256, 0, stream>>>(WoutB, Ht, b_out, x, out);
}
